// LiquidRNN_11458972746383
// MI455X (gfx1250) — matrix operand order repaired and checked by reading, not compiled, not run on hardware
//
#include <hip/hip_runtime.h>
#include <math.h>

// ---------------- problem constants ----------------
#define BB   32
#define TT   1024
#define INS  512
#define SEN  512
#define HID  1024
#define MOT  512
#define OUTS 512
#define NBLK 16          // persistent blocks in the scan kernel

typedef __attribute__((ext_vector_type(16))) _Float16 v16h;
typedef __attribute__((ext_vector_type(8)))  _Float16 v8h;
typedef __attribute__((ext_vector_type(8)))  float    v8f;

// ---- WMMA helper: D = A(16x32 f16) * B(32x16 f16) + C(16x16 f32) ----
__device__ __forceinline__ v8f wmma_f16(v16h a, v16h b, v8f c) {
    return __builtin_amdgcn_wmma_f32_16x16x32_f16(
        /*neg_a=*/false, a, /*neg_b=*/false, b,
        /*c_mod=*/(short)0, c, /*reuse_a=*/false, /*reuse_b=*/false);
}

// A-operand assembly, contiguous K order (matches B):
// lane L, halves 0..7  hold K = k0 + (L<16?0:16) + j
// lane L, halves 8..15 hold K = k0 + (L<16?8:24) + (j-8)
__device__ __forceinline__ v16h load_a_f16(const _Float16* __restrict__ p,
                                           int ao1, int ao2) {
    v8h lo = *(const v8h*)(p + ao1);
    v8h hi = *(const v8h*)(p + ao2);
    v16h a;
#pragma unroll
    for (int i = 0; i < 8; ++i) { a[i] = lo[i]; a[i + 8] = hi[i]; }
    return a;
}

__device__ __forceinline__ v16h load_a_f32(const float* __restrict__ p,
                                           int ao1, int ao2) {
    v16h a;
#pragma unroll
    for (int i = 0; i < 8; ++i) {
        a[i]     = (_Float16)p[ao1 + i];
        a[i + 8] = (_Float16)p[ao2 + i];
    }
    return a;
}

// ---------------- kernel 0: f32 -> f16 weight conversion ----------------
__global__ void __launch_bounds__(256)
cvt_f32_f16_kernel(const float* __restrict__ src, _Float16* __restrict__ dst, int n) {
    int i = blockIdx.x * 256 + threadIdx.x;
    if (i < n) dst[i] = (_Float16)src[i];
}

// ---------------- kernel 1: emb = inputs @ W_in^T + b_in (f16 out) --------
// M = B*T = 32768 rows (b-major, matching inputs layout), N = SEN, K = INS.
__global__ void __launch_bounds__(256)
embed_kernel(const float* __restrict__ x,       // (B,T,INS) f32
             const _Float16* __restrict__ Win,  // (SEN,INS) f16
             const float* __restrict__ b_in,    // (SEN)
             _Float16* __restrict__ emb)        // (B*T, SEN) f16
{
    const int lane  = threadIdx.x & 31;
    const int lhalf = lane >> 4;
    const int lmod  = lane & 15;
    const int ao1 = lhalf ? 16 : 0, ao2 = lhalf ? 24 : 8, bko = lhalf ? 16 : 0;

    const int wave = threadIdx.x >> 5;
    const int tile = blockIdx.x * 8 + wave;   // 2048 m-tiles * 32 n-tiles
    const int m0 = (tile >> 5) * 16;
    const int n0 = (tile & 31) * 16;
    const int n  = n0 + lmod;

    const float*    arow = x   + (size_t)(m0 + lmod) * INS;
    const _Float16* brow = Win + (size_t)n * INS;

    v8f c = {};
    for (int ks = 0; ks < INS; ks += 32) {
        v16h a = load_a_f32(arow + ks, ao1, ao2);
        v16h b = *(const v16h*)(brow + ks + bko);
        c = wmma_f16(a, b, c);
    }
    const float bias = b_in[n];
#pragma unroll
    for (int r = 0; r < 8; ++r) {
        int gr = m0 + (lhalf ? r + 8 : r);
        emb[(size_t)gr * SEN + n] = (_Float16)(c[r] + bias);
    }
}

// ---------------- kernel 2: persistent recurrent scan ----------------
// 16 blocks x 256 threads; block owns 64 hidden columns; wave owns one
// 16x16 tile of the (32 x 1024) pre-activation. Double-buffered f16 h,
// one device-scope release/acquire barrier per timestep.
__device__ __forceinline__ void grid_sync(int* bar, int it) {
    __threadfence();
    __syncthreads();
    if (threadIdx.x == 0) {
        __hip_atomic_fetch_add(bar, 1, __ATOMIC_RELEASE, __HIP_MEMORY_SCOPE_AGENT);
        const int target = NBLK * (it + 1);
        while (__hip_atomic_load(bar, __ATOMIC_ACQUIRE, __HIP_MEMORY_SCOPE_AGENT) < target) {
            __builtin_amdgcn_s_sleep(2);
        }
    }
    __syncthreads();
    __threadfence();
}

__global__ void __launch_bounds__(256)
scan_kernel(const _Float16* __restrict__ emb,   // (B*T, SEN) f16, row = b*T+t
            const float* __restrict__ tsp,      // (B,T)
            const float* __restrict__ init_h,   // (B,HID)
            const _Float16* __restrict__ Wx,    // (HID,SEN) f16
            const _Float16* __restrict__ Wh,    // (HID,HID) f16
            const float* __restrict__ b_cell,   // (HID)
            const float* __restrict__ tau,      // (HID)
            const float* __restrict__ Avec,     // (HID)
            _Float16* __restrict__ hbuf,        // 2 * (B*HID) f16 (double buffer)
            _Float16* __restrict__ hseq,        // (T*B, MOT) f16
            float* __restrict__ last_out,       // (B,HID) f32 (tail of d_out)
            int* __restrict__ bar)
{
    const int lane  = threadIdx.x & 31;
    const int lhalf = lane >> 4;
    const int lmod  = lane & 15;
    const int ao1 = lhalf ? 16 : 0, ao2 = lhalf ? 24 : 8, bko = lhalf ? 16 : 0;

    const int wave = threadIdx.x >> 5;
    const int m0 = (wave >> 2) * 16;                 // batch tile (0 or 16)
    const int n0 = blockIdx.x * 64 + (wave & 3) * 16;
    const int n  = n0 + lmod;                        // this lane's hidden column

    const float bcn = b_cell[n];
    const float An  = Avec[n];
    const float itn = 1.0f / tau[n];
    const bool  is_motor = (n0 < MOT);               // uniform per block

    const int ba = m0 + lmod;                        // A-operand batch row
    const _Float16* wxrow = Wx + (size_t)n * SEN;
    const _Float16* whrow = Wh + (size_t)n * HID;

    int   brow[8];
    float hprev[8];
#pragma unroll
    for (int r = 0; r < 8; ++r) {
        brow[r]  = m0 + (lhalf ? r + 8 : r);
        hprev[r] = init_h[(size_t)brow[r] * HID + n];
        hbuf[(size_t)brow[r] * HID + n] = (_Float16)hprev[r];   // buffer 0
    }

    for (int t = 0; t < TT; ++t) {
        grid_sync(bar, t);   // all writes of step t-1 (buffer t&1) visible

        const _Float16* hrd = hbuf + (size_t)(t & 1) * (BB * HID);
        _Float16*       hwr = hbuf + (size_t)((t + 1) & 1) * (BB * HID);

        v8f c = {};
        // x @ W_x^T   (K = SEN)
        const _Float16* arow = emb + ((size_t)ba * TT + t) * SEN;
        for (int ks = 0; ks < SEN; ks += 32) {
            v16h a = load_a_f16(arow + ks, ao1, ao2);
            v16h b = *(const v16h*)(wxrow + ks + bko);
            c = wmma_f16(a, b, c);
        }
        // h @ W_h^T   (K = HID)
        const _Float16* hrow = hrd + (size_t)ba * HID;
        for (int ks = 0; ks < HID; ks += 32) {
            v16h a = load_a_f16(hrow + ks, ao1, ao2);
            v16h b = *(const v16h*)(whrow + ks + bko);
            c = wmma_f16(a, b, c);
        }
        // elementwise liquid update
#pragma unroll
        for (int r = 0; r < 8; ++r) {
            const int b = brow[r];
            float pre = c[r] + bcn;
            float f   = 1.0f / (1.0f + __expf(-pre));
            float ts  = tsp[(size_t)b * TT + t];
            float hn  = (hprev[r] + ts * f * An) / (1.0f + ts * (itn + f));
            hprev[r] = hn;
            hwr[(size_t)b * HID + n] = (_Float16)hn;
            if (is_motor)
                hseq[((size_t)t * BB + b) * MOT + n] = (_Float16)hn;
        }
    }
    // final state (own cells, written from registers)
#pragma unroll
    for (int r = 0; r < 8; ++r)
        last_out[(size_t)brow[r] * HID + n] = hprev[r];
}

// ---------------- kernel 3: out = motor_seq @ W_out^T + b_out ------------
// M rows are (t*B + b) as produced by the scan; output scattered to (B,T,O).
__global__ void __launch_bounds__(256)
out_kernel(const _Float16* __restrict__ hseq,   // (T*B, MOT) f16
           const _Float16* __restrict__ Wout,   // (OUTS,MOT) f16
           const float* __restrict__ b_out,     // (OUTS)
           float* __restrict__ out)             // (B,T,OUTS) f32
{
    const int lane  = threadIdx.x & 31;
    const int lhalf = lane >> 4;
    const int lmod  = lane & 15;
    const int ao1 = lhalf ? 16 : 0, ao2 = lhalf ? 24 : 8, bko = lhalf ? 16 : 0;

    const int wave = threadIdx.x >> 5;
    const int tile = blockIdx.x * 8 + wave;     // 2048 m-tiles * 32 n-tiles
    const int m0 = (tile >> 5) * 16;
    const int n0 = (tile & 31) * 16;
    const int n  = n0 + lmod;

    const _Float16* arow = hseq + (size_t)(m0 + lmod) * MOT;
    const _Float16* brow = Wout + (size_t)n * MOT;

    v8f c = {};
    for (int ks = 0; ks < MOT; ks += 32) {
        v16h a = load_a_f16(arow + ks, ao1, ao2);
        v16h b = *(const v16h*)(brow + ks + bko);
        c = wmma_f16(a, b, c);
    }
    const float bias = b_out[n];
#pragma unroll
    for (int r = 0; r < 8; ++r) {
        int gr = m0 + (lhalf ? r + 8 : r);
        int t = gr >> 5, b = gr & 31;
        out[((size_t)b * TT + t) * OUTS + n] = c[r] + bias;
    }
}

// ---------------- host launcher ----------------
extern "C" void kernel_launch(void* const* d_in, const int* in_sizes, int n_in,
                              void* d_out, int out_size, void* d_ws, size_t ws_size,
                              hipStream_t stream) {
    const float* inputs    = (const float*)d_in[0];
    const float* timespans = (const float*)d_in[1];
    const float* init_h    = (const float*)d_in[2];
    const float* W_in      = (const float*)d_in[3];
    const float* b_in      = (const float*)d_in[4];
    const float* W_x       = (const float*)d_in[5];
    const float* W_h       = (const float*)d_in[6];
    const float* b_cell    = (const float*)d_in[7];
    const float* tau       = (const float*)d_in[8];
    const float* A         = (const float*)d_in[9];
    const float* W_out     = (const float*)d_in[10];
    const float* b_out     = (const float*)d_in[11];

    float* out      = (float*)d_out;
    float* last_out = out + (size_t)BB * TT * OUTS;

    // workspace carve-up (256B aligned)
    char* ws = (char*)d_ws;
    size_t off = 0;
    auto carve = [&](size_t bytes) {
        void* p = ws + off;
        off += (bytes + 255) & ~(size_t)255;
        return p;
    };
    _Float16* Win16  = (_Float16*)carve((size_t)SEN * INS * 2);
    _Float16* Wx16   = (_Float16*)carve((size_t)HID * SEN * 2);
    _Float16* Wh16   = (_Float16*)carve((size_t)HID * HID * 2);
    _Float16* Wout16 = (_Float16*)carve((size_t)OUTS * MOT * 2);
    _Float16* emb16  = (_Float16*)carve((size_t)BB * TT * SEN * 2);
    _Float16* hseq16 = (_Float16*)carve((size_t)TT * BB * MOT * 2);
    _Float16* hbuf   = (_Float16*)carve((size_t)2 * BB * HID * 2);
    int*      bar    = (int*)carve(256);

    hipMemsetAsync(bar, 0, 256, stream);

    auto cvt = [&](const float* s, _Float16* d, int n) {
        cvt_f32_f16_kernel<<<(n + 255) / 256, 256, 0, stream>>>(s, d, n);
    };
    cvt(W_in,  Win16,  SEN * INS);
    cvt(W_x,   Wx16,   HID * SEN);
    cvt(W_h,   Wh16,   HID * HID);
    cvt(W_out, Wout16, OUTS * MOT);

    // embed: 2048 m-tiles * 32 n-tiles / 8 waves-per-block
    embed_kernel<<<8192, 256, 0, stream>>>(inputs, Win16, b_in, emb16);

    // persistent scan: 16 co-resident blocks
    scan_kernel<<<NBLK, 256, 0, stream>>>(emb16, timespans, init_h, Wx16, Wh16,
                                          b_cell, tau, A, hbuf, hseq16,
                                          last_out, bar);

    // output projection
    out_kernel<<<8192, 256, 0, stream>>>(hseq16, Wout16, b_out, out);
}
